// EnhancedGNNModelHousehold_11115375362930
// MI455X (gfx1250) — matrix operand order repaired and checked by reading, not compiled, not run on hardware
//
#include <hip/hip_runtime.h>
#include <hip/hip_bf16.h>

// ---------------- problem constants (from reference) ----------------
#define N_NODES  300000
#define NH_NODES 250000
#define IN_CH    32
#define HID      128
#define MLP_HID  256
#define NEDGE    2000000
#define EPS_V    1e-5f

typedef __attribute__((ext_vector_type(16))) __bf16 v16bf;
typedef __attribute__((ext_vector_type(8)))  float  v8f;

// fp32 -> bf16 round-to-nearest-even
static __device__ __forceinline__ __bf16 f2bf(float f) {
    union { float f; unsigned u; } v; v.f = f;
    unsigned r = v.u + 0x7FFFu + ((v.u >> 16) & 1u);
    unsigned short h = (unsigned short)(r >> 16);
    return __builtin_bit_cast(__bf16, h);
}

// ---------------- CDNA5 async global->LDS copy (ASYNCcnt path) ----------------
// the LDS address comes from the VDST VGPR; for generic shared pointers the LDS offset is
// addr[31:0] (ISA 10.2 aperture rules), so truncating the flat address works.
static __device__ __forceinline__ unsigned lds_off32(const void* p) {
    return (unsigned)(size_t)p;
}
static __device__ __forceinline__ void async_ld_b128(unsigned lds_addr, const void* gaddr) {
    asm volatile("global_load_async_to_lds_b128 %0, %1, off"
                 :: "v"(lds_addr), "v"(gaddr)
                 : "memory");
}
static __device__ __forceinline__ void wait_async0() {
    asm volatile("s_wait_asynccnt 0" ::: "memory");
}

// ---------------- WMMA fragment loaders (contiguous k order, same for A and B) ----------------
// A-matrix 16x32 bf16: lane L(0-15)=row M=L holds K = h*16+e (e<16),
// where h = lane>>4.  tile = &A[row0*lda + k0]
static __device__ __forceinline__ v16bf load_a_frag(const __bf16* __restrict__ tile,
                                                    int lda, int lane) {
    const int m = lane & 15, h = lane >> 4;
    const __bf16* p = tile + (long)m * lda + h * 16;
    v16bf a;
#pragma unroll
    for (int e = 0; e < 8; ++e) a[e] = p[e];
#pragma unroll
    for (int e = 0; e < 8; ++e) a[8 + e] = p[8 + e];
    return a;
}

// B-matrix 32x16 bf16 where B = W^T, W row-major [CO][K]:
// lane: n = n0+(lane&15), element e holds K = k0 + (lane>>4)*16 + e  -> 16
// contiguous bf16 from one row of W.
static __device__ __forceinline__ v16bf load_b_frag(const __bf16* __restrict__ W,
                                                    int ldw, int n0, int k0, int lane) {
    const __bf16* p = W + (long)(n0 + (lane & 15)) * ldw + k0 + (lane >> 4) * 16;
    v16bf b;
#pragma unroll
    for (int e = 0; e < 16; ++e) b[e] = p[e];
    return b;
}

static __device__ __forceinline__ v8f wmma_bf16(v16bf a, v16bf b, v8f c) {
    return __builtin_amdgcn_wmma_f32_16x16x32_bf16(false, a, false, b,
                                                   (short)0, c, false, false);
}

// ---------------- small utility kernels ----------------
__global__ __launch_bounds__(256) void f32_to_bf16_kernel(const float* __restrict__ in,
                                                          __bf16* __restrict__ out, long n) {
    long i = (long)blockIdx.x * blockDim.x + threadIdx.x;
    if (i < n) out[i] = f2bf(in[i]);
}

// Zero-padded bf16 copy of a head W2 [co x 256] into [16 x 256]
__global__ __launch_bounds__(256) void pad_w2_kernel(const float* __restrict__ W2,
                                                     __bf16* __restrict__ out, int co) {
    int i = blockIdx.x * blockDim.x + threadIdx.x;   // 16*256
    if (i >= 16 * MLP_HID) return;
    int r = i >> 8, k = i & 255;
    out[i] = (r < co) ? f2bf(W2[r * MLP_HID + k]) : f2bf(0.0f);
}

__global__ __launch_bounds__(256) void deg_count_kernel(const long long* __restrict__ dst,
                                                        float* __restrict__ deg) {
    long e = (long)blockIdx.x * blockDim.x + threadIdx.x;
    if (e < NEDGE) atomicAdd(&deg[dst[e]], 1.0f);
}

// ---------------- edge aggregation: agg[dst] += h[src] ----------------
__global__ __launch_bounds__(256) void edge_scatter_kernel(const long long* __restrict__ src,
                                                           const long long* __restrict__ dst,
                                                           const float* __restrict__ h,
                                                           float* __restrict__ agg, int C) {
    const int C4 = C >> 2;
    long t = (long)blockIdx.x * blockDim.x + threadIdx.x;
    if (t >= (long)NEDGE * C4) return;
    long e = t / C4;
    int  c = (int)(t - e * C4) << 2;
    __builtin_prefetch(src + e + 2048, 0, 0);   // global_prefetch_b8
    __builtin_prefetch(dst + e + 2048, 0, 0);
    long s = src[e] * (long)C + c;
    long d = dst[e] * (long)C + c;
    float4 v = *(const float4*)(h + s);
    atomicAdd(&agg[d + 0], v.x);
    atomicAdd(&agg[d + 1], v.y);
    atomicAdd(&agg[d + 2], v.z);
    atomicAdd(&agg[d + 3], v.w);
}

// agg /= max(deg,1), and emit bf16 copies of agg and h
__global__ __launch_bounds__(256) void div_convert_kernel(const float* __restrict__ agg,
                                                          const float* __restrict__ h,
                                                          const float* __restrict__ deg,
                                                          __bf16* __restrict__ aggb,
                                                          __bf16* __restrict__ hb,
                                                          long n, int C) {
    long i = (long)blockIdx.x * blockDim.x + threadIdx.x;
    if (i >= n) return;
    long row = i / C;
    float d = fmaxf(deg[row], 1.0f);
    aggb[i] = f2bf(agg[i] / d);
    hb[i]   = f2bf(h[i]);
}

// ---------------- SAGE GEMM: y = aggb@Wl^T + bl + hb@Wr^T  (CO = 128) ----------------
// A tiles (16 x CI of agg and h) are async-staged to LDS once per block; all 8
// waves then feed their WMMAs from ds_load while B (weights) streams from L2.
template <int CI>
__global__ __launch_bounds__(256) void sage_gemm_kernel(const __bf16* __restrict__ aggb,
                                                        const __bf16* __restrict__ hb,
                                                        const __bf16* __restrict__ Wlb,
                                                        const __bf16* __restrict__ Wrb,
                                                        const float* __restrict__ bl,
                                                        float* __restrict__ y) {
    __shared__ __bf16 sAgg[16 * CI];
    __shared__ __bf16 sH[16 * CI];
    const int lane = threadIdx.x & 31;
    const int wave = threadIdx.x >> 5;
    const long row0 = (long)blockIdx.x * 16;

    constexpr int NLD = (16 * CI * 2) / 16;       // one b128 per participating lane
    const char* gAgg = (const char*)(aggb + row0 * CI);
    const char* gH   = (const char*)(hb + row0 * CI);
    if (threadIdx.x < NLD) {
        const unsigned o = threadIdx.x * 16;
        async_ld_b128(lds_off32(sAgg) + o, gAgg + o);
        async_ld_b128(lds_off32(sH) + o, gH + o);
    }
    wait_async0();
    __syncthreads();

    const int n0 = wave * 16;                     // 8 waves cover 128 output cols
    v8f acc = {};
#pragma unroll
    for (int k0 = 0; k0 < CI; k0 += 32) {
        v16bf a0 = load_a_frag(sAgg + k0, CI, lane);
        v16bf b0 = load_b_frag(Wlb, CI, n0, k0, lane);
        acc = wmma_bf16(a0, b0, acc);
        v16bf a1 = load_a_frag(sH + k0, CI, lane);
        v16bf b1 = load_b_frag(Wrb, CI, n0, k0, lane);
        acc = wmma_bf16(a1, b1, acc);
    }
    const int col = n0 + (lane & 15);
    const long r0 = row0 + (lane >> 4) * 8;
    const float bias = bl[col];
#pragma unroll
    for (int r = 0; r < 8; ++r)
        y[(r0 + r) * HID + col] = acc[r] + bias;
}

// ---------------- GraphNorm reductions ----------------
__global__ __launch_bounds__(HID) void col_sum_kernel(const float* __restrict__ y,
                                                      float* __restrict__ colsum, int nrows) {
    int c = threadIdx.x;                       // 128 threads = 128 feature cols
    long r0 = (long)blockIdx.x * 256;
    long re = r0 + 256; if (re > nrows) re = nrows;
    float s = 0.0f;
    for (long r = r0; r < re; ++r) s += y[r * HID + c];
    atomicAdd(&colsum[c], s);
}

__global__ __launch_bounds__(HID) void col_sqsum_kernel(const float* __restrict__ y,
                                                        const float* __restrict__ colsum,
                                                        const float* __restrict__ alpha,
                                                        float* __restrict__ sqsum, int nrows) {
    int c = threadIdx.x;
    const float mu = colsum[c] * (1.0f / (float)N_NODES);
    const float am = alpha[c] * mu;
    long r0 = (long)blockIdx.x * 256;
    long re = r0 + 256; if (re > nrows) re = nrows;
    float s = 0.0f;
    for (long r = r0; r < re; ++r) { float cc = y[r * HID + c] - am; s += cc * cc; }
    atomicAdd(&sqsum[c], s);
}

__global__ __launch_bounds__(256) void norm_relu_kernel(const float* __restrict__ y,
                                                        const float* __restrict__ colsum,
                                                        const float* __restrict__ sqsum,
                                                        const float* __restrict__ w,
                                                        const float* __restrict__ b,
                                                        const float* __restrict__ alpha,
                                                        float* __restrict__ hout, long n) {
    long i = (long)blockIdx.x * blockDim.x + threadIdx.x;
    if (i >= n) return;
    int c = (int)(i & (HID - 1));
    float mu  = colsum[c] * (1.0f / (float)N_NODES);
    float var = sqsum[c]  * (1.0f / (float)N_NODES);
    float cc = y[i] - alpha[c] * mu;
    float o = w[c] * cc * rsqrtf(var + EPS_V) + b[c];
    hout[i] = fmaxf(o, 0.0f);
}

// ---------------- fused 3-head MLP ----------------
struct MlpArgs {
    const __bf16* hb;            // NH x 128 bf16
    const __bf16* W1b;           // 3 heads x [256 x 128] bf16
    const __bf16* W2b;           // 3 heads x [16 x 256] bf16 (zero padded)
    const float*  b1[3];
    const float*  b2[3];
    float*        out[3];
    int           co[3];
};

__global__ __launch_bounds__(256) void mlp_heads_kernel(MlpArgs A) {
    __shared__ __bf16 sX[16 * HID];            // async-staged input tile (4 KB)
    __shared__ __bf16 z[3][16][MLP_HID];       // 24 KB, relu(h@W1^T+b1) per head, bf16
    const int lane = threadIdx.x & 31;
    const int wave = threadIdx.x >> 5;
    const long row0 = (long)blockIdx.x * 16;

    // async global->LDS copy of the 16x128 input tile (one b128 per thread)
    {
        const char* gX = (const char*)(A.hb + row0 * HID);
        const unsigned o = threadIdx.x * 16;
        async_ld_b128(lds_off32(sX) + o, gX + o);
        wait_async0();
        __syncthreads();
    }

    // ---- stage 1: three 16x256 hidden activations (48 wave-tiles / 8 waves) ----
    for (int t = wave; t < 48; t += 8) {
        const int head = t >> 4;
        const int n0 = (t & 15) * 16;
        const __bf16* W1h = A.W1b + (long)head * MLP_HID * HID;
        v8f acc = {};
#pragma unroll
        for (int k0 = 0; k0 < HID; k0 += 32) {
            v16bf a = load_a_frag(sX + k0, HID, lane);
            v16bf b = load_b_frag(W1h, HID, n0, k0, lane);
            acc = wmma_bf16(a, b, acc);
        }
        const int col = n0 + (lane & 15);
        const int r0 = (lane >> 4) * 8;
        const float bias = A.b1[head][col];
#pragma unroll
        for (int r = 0; r < 8; ++r)
            z[head][r0 + r][col] = f2bf(fmaxf(acc[r] + bias, 0.0f));
    }
    __syncthreads();

    // ---- stage 2: head outputs 16 x co (waves 0..2, K = 256 from LDS) ----
    if (wave < 3) {
        const int head = wave;
        const __bf16* W2h = A.W2b + (long)head * 16 * MLP_HID;
        v8f acc = {};
#pragma unroll
        for (int k0 = 0; k0 < MLP_HID; k0 += 32) {
            v16bf a = load_a_frag(&z[head][0][0] + k0, MLP_HID, lane);  // ds_load path
            v16bf b = load_b_frag(W2h, MLP_HID, 0, k0, lane);
            acc = wmma_bf16(a, b, acc);
        }
        const int col = lane & 15;
        const int r0 = (lane >> 4) * 8;
        const int co = A.co[head];
        if (col < co) {
            const float bias = A.b2[head][col];
            float* op = A.out[head];
#pragma unroll
            for (int r = 0; r < 8; ++r)
                op[(row0 + r0 + r) * co + col] = acc[r] + bias;
        }
    }
}

// ---------------- host orchestration ----------------
extern "C" void kernel_launch(void* const* d_in, const int* in_sizes, int n_in,
                              void* d_out, int out_size, void* d_ws, size_t ws_size,
                              hipStream_t stream) {
    (void)in_sizes; (void)n_in; (void)out_size; (void)ws_size;

    const float*     x   = (const float*)d_in[0];
    const long long* src = (const long long*)d_in[1];
    const long long* dst = (const long long*)d_in[2];

    const float *Wl[3], *bl[3], *Wr[3], *nw[3], *nb[3], *na[3];
    int idx = 3;
    for (int l = 0; l < 3; ++l) {
        Wl[l] = (const float*)d_in[idx + 0];
        bl[l] = (const float*)d_in[idx + 1];
        Wr[l] = (const float*)d_in[idx + 2];
        nw[l] = (const float*)d_in[idx + 3];
        nb[l] = (const float*)d_in[idx + 4];
        na[l] = (const float*)d_in[idx + 5];
        idx += 6;
    }
    const float *W1[3], *b1[3], *W2[3], *b2[3];
    for (int h = 0; h < 3; ++h) {
        W1[h] = (const float*)d_in[idx + 0];
        b1[h] = (const float*)d_in[idx + 1];
        W2[h] = (const float*)d_in[idx + 2];
        b2[h] = (const float*)d_in[idx + 3];
        idx += 4;
    }

    // workspace carve-out
    size_t off = 0;
    auto alloc = [&](size_t bytes) -> void* {
        off = (off + 255) & ~(size_t)255;
        void* p = (char*)d_ws + off;
        off += bytes;
        return p;
    };
    float*  deg    = (float*)alloc((size_t)N_NODES * 4);
    float*  hcur   = (float*)alloc((size_t)N_NODES * HID * 4);
    float*  ybuf   = (float*)alloc((size_t)N_NODES * HID * 4);
    float*  agg    = (float*)alloc((size_t)N_NODES * HID * 4);
    __bf16* hb     = (__bf16*)alloc((size_t)N_NODES * HID * 2);
    __bf16* aggb   = (__bf16*)alloc((size_t)N_NODES * HID * 2);
    float*  colsum = (float*)alloc(HID * 4);
    float*  sqsum  = (float*)alloc(HID * 4);
    __bf16* Wlb[3], *Wrb[3];
    for (int l = 0; l < 3; ++l) {
        Wlb[l] = (__bf16*)alloc((size_t)HID * HID * 2);
        Wrb[l] = (__bf16*)alloc((size_t)HID * HID * 2);
    }
    __bf16* W1b = (__bf16*)alloc((size_t)3 * MLP_HID * HID * 2);
    __bf16* W2b = (__bf16*)alloc((size_t)3 * 16 * MLP_HID * 2);

    // weight conversions (bf16)
    for (int l = 0; l < 3; ++l) {
        long nW = (long)HID * (l == 0 ? IN_CH : HID);
        f32_to_bf16_kernel<<<(nW + 255) / 256, 256, 0, stream>>>(Wl[l], Wlb[l], nW);
        f32_to_bf16_kernel<<<(nW + 255) / 256, 256, 0, stream>>>(Wr[l], Wrb[l], nW);
    }
    const int co_arr[3] = {8, 5, 9};
    for (int h = 0; h < 3; ++h) {
        long nW1 = (long)MLP_HID * HID;
        f32_to_bf16_kernel<<<(nW1 + 255) / 256, 256, 0, stream>>>(W1[h], W1b + (long)h * nW1, nW1);
        pad_w2_kernel<<<(16 * MLP_HID + 255) / 256, 256, 0, stream>>>(W2[h], W2b + (long)h * 16 * MLP_HID, co_arr[h]);
    }

    // degrees
    hipMemsetAsync(deg, 0, (size_t)N_NODES * 4, stream);
    deg_count_kernel<<<(NEDGE + 255) / 256, 256, 0, stream>>>(dst, deg);

    // 3 SAGE + GraphNorm + ReLU layers
    for (int l = 0; l < 3; ++l) {
        const int ci = (l == 0) ? IN_CH : HID;
        const float* hin = (l == 0) ? x : hcur;

        hipMemsetAsync(agg, 0, (size_t)N_NODES * ci * 4, stream);
        long st = (long)NEDGE * (ci / 4);
        edge_scatter_kernel<<<(st + 255) / 256, 256, 0, stream>>>(src, dst, hin, agg, ci);

        long nel = (long)N_NODES * ci;
        div_convert_kernel<<<(nel + 255) / 256, 256, 0, stream>>>(agg, hin, deg, aggb, hb, nel, ci);

        if (l == 0)
            sage_gemm_kernel<IN_CH><<<N_NODES / 16, 256, 0, stream>>>(aggb, hb, Wlb[0], Wrb[0], bl[0], ybuf);
        else
            sage_gemm_kernel<HID><<<N_NODES / 16, 256, 0, stream>>>(aggb, hb, Wlb[l], Wrb[l], bl[l], ybuf);

        hipMemsetAsync(colsum, 0, HID * 4, stream);
        hipMemsetAsync(sqsum, 0, HID * 4, stream);
        col_sum_kernel<<<(N_NODES + 255) / 256, HID, 0, stream>>>(ybuf, colsum, N_NODES);
        col_sqsum_kernel<<<(N_NODES + 255) / 256, HID, 0, stream>>>(ybuf, colsum, na[l], sqsum, N_NODES);
        long nall = (long)N_NODES * HID;
        norm_relu_kernel<<<(nall + 255) / 256, 256, 0, stream>>>(ybuf, colsum, sqsum, nw[l], nb[l], na[l], hcur, nall);
    }

    // bf16 copy of final features for the MLP heads (first NH rows)
    long nhh = (long)NH_NODES * HID;
    f32_to_bf16_kernel<<<(nhh + 255) / 256, 256, 0, stream>>>(hcur, hb, nhh);

    MlpArgs MA;
    MA.hb = hb; MA.W1b = W1b; MA.W2b = W2b;
    float* out_hh  = (float*)d_out;
    float* out_eth = out_hh + (long)NH_NODES * 8;
    float* out_rel = out_eth + (long)NH_NODES * 5;
    float* outs[3] = {out_hh, out_eth, out_rel};
    for (int h = 0; h < 3; ++h) {
        MA.b1[h] = b1[h]; MA.b2[h] = b2[h];
        MA.out[h] = outs[h]; MA.co[h] = co_arr[h];
    }
    mlp_heads_kernel<<<NH_NODES / 16, 256, 0, stream>>>(MA);
}
